// TFLSHSelfAttention_68831145886362
// MI455X (gfx1250) — compile-verified, matrix operand order checked by reading, not run on hardware
//
#include <hip/hip_runtime.h>
#include <hip/hip_bf16.h>
#include <math.h>

#define EMB    1024
#define HEADS  8
#define DH     128
#define BATCH  2
#define TLEN   8192
#define BH     (BATCH*HEADS)      // 16
#define NHASH  8
#define BUCKET 64
#define NBUCK  (TLEN/BUCKET)      // 128
#define NCHUNK (NHASH*NBUCK)      // 1024

#define USE_ASYNC_LDS 1

typedef __attribute__((ext_vector_type(16))) __bf16 v16bf;
typedef __attribute__((ext_vector_type(8)))  float  v8f;

__device__ __forceinline__ v8f wmma_bf16(v16bf a, v16bf b, v8f c) {
  return __builtin_amdgcn_wmma_f32_16x16x32_bf16(false, a, false, b, (short)0, c,
                                                 false, false);
}

__device__ __forceinline__ v8f zero8() {
  v8f z = {0.f, 0.f, 0.f, 0.f, 0.f, 0.f, 0.f, 0.f};
  return z;
}

// Per-lane 16-byte global -> LDS copy.
// Async path: GLOBAL_LOAD_ASYNC_TO_LDS_B128 (tracked with ASYNCcnt).
// lds_ptr: generic pointer to the LDS destination (for fallback);
// lds_off: raw LDS byte offset (low 32 bits of the generic LDS address).
__device__ __forceinline__ void copy16_to_lds(void* lds_ptr, unsigned lds_off,
                                              const void* g) {
#if USE_ASYNC_LDS
  asm volatile("global_load_async_to_lds_b128 %0, %1, off"
               :: "v"(lds_off), "v"(g)
               : "memory");
#else
  *(uint4*)lds_ptr = *(const uint4*)g;
#endif
}

__device__ __forceinline__ void async_copies_wait() {
#if USE_ASYNC_LDS
  asm volatile("s_wait_asynccnt 0x0" ::: "memory");
#endif
}

// Load one 16x32 bf16 WMMA fragment from a row-major LDS tile.
// Per CDNA5 ISA layout: lane (m = lane&15, hi = lane>>4) holds, for element e,
// K = hi*8 + e (e<8) and K = hi*8 + 16 + (e-8) (e>=8): two contiguous 8-elem runs.
// For the B operand use a tile stored transposed (Bt[n][k]); then the same
// loader applies with row = n-base.
__device__ __forceinline__ v16bf load_frag(const __bf16* base, int row, int ldr,
                                           int koff, int lane) {
  int m = lane & 15, hi = lane >> 4;
  const __bf16* p = base + (size_t)(row + m) * ldr + koff + hi * 8;
  v16bf f;
#pragma unroll
  for (int e = 0; e < 8; ++e) f[e] = p[e];
#pragma unroll
  for (int e = 0; e < 8; ++e) f[8 + e] = p[16 + e];
  return f;
}

// ---------------------------------------------------------------- cvt f32->bf16
__global__ void cvt_bf16_kernel(const float* __restrict__ in,
                                __bf16* __restrict__ out, int n) {
  int i = blockIdx.x * blockDim.x + threadIdx.x;
  if (i < n) out[i] = (__bf16)in[i];
}

// ------------------------------------- cvt f32->bf16 with transpose (weights)
// in: [K][N] row-major f32 ; out: [N][K] row-major bf16
__global__ void cvt_bf16_t_kernel(const float* __restrict__ in,
                                  __bf16* __restrict__ out, int Nn, int Kk) {
  int i = blockIdx.x * blockDim.x + threadIdx.x;
  if (i >= Nn * Kk) return;
  int n = i / Kk, k = i % Kk;
  out[i] = (__bf16)in[(size_t)k * Nn + n];
}

// ------------------------------------------------------------------- WMMA GEMM
// C[M,N] (f32) = A[M,K] (bf16, row-major) * Bt[N,K]^T (bf16, row-major) [+ bias]
// Both operands are row-major over K -> both tiles stage as contiguous rows via
// async b128 global->LDS copies. Block tile 128x128, K-step 32, 8 waves.
__global__ __launch_bounds__(256) void gemm_bf16_kernel(
    const __bf16* __restrict__ A, const __bf16* __restrict__ Bt,
    float* __restrict__ C, const float* __restrict__ bias,
    int M, int N, int K) {
  __shared__ __bf16 As[128 * 32];    // As[m][k]
  __shared__ __bf16 Bts[128 * 32];   // Bts[n][k]
  int tid = threadIdx.x, lane = tid & 31, w = tid >> 5;
  int mbase = blockIdx.y * 128, nbase = blockIdx.x * 128;
  int wm = (w & 3) * 32, wn = (w >> 2) * 64;
  unsigned as_off = (unsigned)(size_t)(void*)As;   // LDS byte offsets
  unsigned bs_off = (unsigned)(size_t)(void*)Bts;

  v8f acc[2][4];
#pragma unroll
  for (int tm = 0; tm < 2; ++tm)
#pragma unroll
    for (int tn = 0; tn < 4; ++tn) acc[tm][tn] = zero8();

  for (int ko = 0; ko < K; ko += 32) {
    __syncthreads();
    // 128 rows x 32 cols = 512 16-byte chunks per tile; 2 chunks/thread/tile.
#pragma unroll
    for (int ch = tid; ch < 512; ch += 256) {
      int r = ch >> 2, cc = (ch & 3) * 8;
      int le = r * 32 + cc;
      copy16_to_lds(As + le, as_off + le * 2,
                    A + (size_t)(mbase + r) * K + ko + cc);
      copy16_to_lds(Bts + le, bs_off + le * 2,
                    Bt + (size_t)(nbase + r) * K + ko + cc);
    }
    async_copies_wait();
    __syncthreads();
#pragma unroll
    for (int tm = 0; tm < 2; ++tm) {
      v16bf af = load_frag(As, wm + tm * 16, 32, 0, lane);
#pragma unroll
      for (int tn = 0; tn < 4; ++tn) {
        v16bf bf = load_frag(Bts, wn + tn * 16, 32, 0, lane);
        acc[tm][tn] = wmma_bf16(af, bf, acc[tm][tn]);
      }
    }
  }

  int hi = lane >> 4, col0 = lane & 15;
#pragma unroll
  for (int tm = 0; tm < 2; ++tm)
#pragma unroll
    for (int tn = 0; tn < 4; ++tn)
#pragma unroll
      for (int r = 0; r < 8; ++r) {
        int row = mbase + wm + tm * 16 + hi * 8 + r;
        int col = nbase + wn + tn * 16 + col0;
        float vv = acc[tm][tn][r];
        if (bias) vv += bias[col];
        C[(size_t)row * N + col] = vv;
      }
}

// ------------------------------------------------------- LSH hashing + norms
// grid = BH * (TLEN/64), block = 64 threads (one token each)
__global__ __launch_bounds__(64) void hash_kernel(
    const float* __restrict__ qk, const float* __restrict__ rot,
    unsigned char* __restrict__ buckets, float* __restrict__ normf) {
  extern __shared__ char smem[];
  float* qrow = (float*)smem;            // [64][128]
  float* rs   = qrow + 64 * 128;         // [128][64]  rs[d*64+i]
  int bh = blockIdx.x >> 7;
  int tc = blockIdx.x & 127;
  int t0 = tc * 64;
  int b = bh >> 3, hh = bh & 7;
  int tid = threadIdx.x;

  for (int i = tid; i < 64 * 128; i += 64) {
    int r = i >> 7, d = i & 127;
    qrow[i] = qk[(size_t)(b * TLEN + t0 + r) * EMB + hh * DH + d];
  }
  __syncthreads();

  float ssq = 0.f;
  for (int d = 0; d < 128; ++d) { float x = qrow[tid * 128 + d]; ssq += x * x; }
  normf[bh * TLEN + t0 + tid] = rsqrtf(fmaxf(ssq, 1e-12f));

  for (int h = 0; h < NHASH; ++h) {
    __syncthreads();
    for (int i = tid; i < 128 * 64; i += 64) {
      int d = i >> 6, ib = i & 63;
      rs[i] = rot[((size_t)d * NHASH + h) * 64 + ib];
    }
    __syncthreads();
    float bestmax = -1e30f, bestmin = 1e30f;
    int imax = 0, imin = 0;
    for (int ib = 0; ib < 64; ++ib) {
      float s = 0.f;
      for (int d = 0; d < 128; ++d) s += qrow[tid * 128 + d] * rs[d * 64 + ib];
      if (s > bestmax) { bestmax = s; imax = ib; }
      if (s < bestmin) { bestmin = s; imin = ib; }
    }
    int bk = (bestmax >= -bestmin) ? imax : 64 + imin;  // argmax over [rv, -rv]
    buckets[((size_t)(bh * NHASH + h)) * TLEN + t0 + tid] = (unsigned char)bk;
  }
}

// ---------------------------------------------- stable counting sort per (bh,h)
// grid = BH*NHASH, block = 128 threads; each thread owns 64 consecutive tokens.
__global__ __launch_bounds__(128) void sort_kernel(
    const unsigned char* __restrict__ buckets, int* __restrict__ st) {
  extern __shared__ char smem[];
  unsigned int* hist   = (unsigned int*)smem;   // [bin][tid] : 128*128
  unsigned int* binsum = hist + 128 * 128;
  unsigned int* binoff = binsum + 128;
  int g = blockIdx.x;
  const unsigned char* bk = buckets + (size_t)g * TLEN;
  int* stg = st + (size_t)g * TLEN;
  int tid = threadIdx.x;

  for (int i = tid; i < 128 * 128; i += 128) hist[i] = 0;
  __syncthreads();
  int t0 = tid * 64;
  for (int j = 0; j < 64; ++j) hist[(int)bk[t0 + j] * 128 + tid]++;  // own column
  __syncthreads();
  unsigned int run = 0;
  for (int c = 0; c < 128; ++c) {
    unsigned int vv = hist[tid * 128 + c];
    hist[tid * 128 + c] = run;
    run += vv;
  }
  binsum[tid] = run;
  __syncthreads();
  if (tid == 0) {
    unsigned int r2 = 0;
    for (int bb = 0; bb < 128; ++bb) { binoff[bb] = r2; r2 += binsum[bb]; }
  }
  __syncthreads();
  for (int c = 0; c < 128; ++c) hist[tid * 128 + c] += binoff[tid];
  __syncthreads();
  for (int j = 0; j < 64; ++j) {      // position order -> stable
    int bb = bk[t0 + j];
    unsigned int r = hist[bb * 128 + tid]++;
    stg[r] = t0 + j;
  }
}

// ------------------------------------------------------- chunked LSH attention
// grid = BH * NCHUNK, block = 128 threads (4 waves, 16 q-rows each)
// phase 0: write per-(bh,h,token) logsumexp. phase 1: weighted PV accumulate.
__global__ __launch_bounds__(128) void lsh_attn_kernel(
    const float* __restrict__ qk, const float* __restrict__ v,
    const float* __restrict__ normf, const int* __restrict__ st,
    const unsigned char* __restrict__ pmask, float* __restrict__ logits,
    const float* __restrict__ wbuf, float* __restrict__ attn, int phase) {
  extern __shared__ char smem[];
  __bf16* Aq    = (__bf16*)smem;             // [64][128]   q * dh^-0.5
  __bf16* Kt    = Aq + 64 * 128;             // [128][128]  normalized keys (key-major)
  __bf16* Vt    = Kt + 128 * 128;            // [128][128]  Vt[d][key]
  float*  dotsS = (float*)(Vt + 128 * 128);  // [64][128]
  __bf16* P     = (__bf16*)(dotsS + 64 * 128); // [64][128] weighted probs
  int*    kid   = (int*)(P + 64 * 128);      // [128] key token ids (first 64 = q ids)

  int bi = blockIdx.x;
  int bh = bi >> 10, c = bi & (NCHUNK - 1);
  int h = c >> 7, cl = c & 127;
  int cp = (c + NCHUNK - 1) & (NCHUNK - 1);   // look-one-back with wrap
  int hp = cp >> 7, clp = cp & 127;
  int b = bh >> 3, hh = bh & 7;
  int tid = threadIdx.x, lane = tid & 31, w = tid >> 5;

  int sidx = (tid < 64) ? ((bh * NHASH + h) * TLEN + cl * 64 + tid)
                        : ((bh * NHASH + hp) * TLEN + clp * 64 + (tid - 64));
  kid[tid] = st[sidx];
  __syncthreads();

  const float scale = 0.08838834764831845f;   // 128^-0.5
  for (int i = tid; i < 64 * 128; i += 128) {
    int r = i >> 7, d = i & 127;
    int tok = kid[r];
    Aq[i] = (__bf16)(qk[(size_t)(b * TLEN + tok) * EMB + hh * DH + d] * scale);
  }
  for (int i = tid; i < 128 * 128; i += 128) {
    int key = i >> 7, d = i & 127;
    int tok = kid[key];
    size_t gi = (size_t)(b * TLEN + tok) * EMB + hh * DH + d;
    Kt[key * 128 + d] = (__bf16)(qk[gi] * normf[bh * TLEN + tok]);
    Vt[d * 128 + key] = (__bf16)(v[gi]);
  }
  __syncthreads();

  int qs = w * 16;
  int hi = lane >> 4, col0 = lane & 15;
  for (int jt = 0; jt < 8; ++jt) {            // dots = Aq @ Kt^T  (64x128)
    v8f acc = zero8();
#pragma unroll
    for (int kk = 0; kk < 4; ++kk) {
      v16bf af = load_frag(Aq, qs, 128, kk * 32, lane);
      v16bf bf = load_frag(Kt, jt * 16, 128, kk * 32, lane);
      acc = wmma_bf16(af, bf, acc);
    }
#pragma unroll
    for (int r = 0; r < 8; ++r) {
      int rowl = qs + hi * 8 + r;
      int col = jt * 16 + col0;
      float dv = acc[r];
      int qi = kid[rowl];
      int ki = kid[col];
      if (pmask[b * TLEN + ki]) dv = -1e9f;   // padding mask
      if (ki == qi) dv = -1e5f;               // self-attention discouraged (last)
      dotsS[rowl * 128 + col] = dv;
    }
  }
  __syncthreads();

  if (tid < 64) {                             // row softmax / lse
    float m = -1e30f;
    for (int col = 0; col < 128; ++col) m = fmaxf(m, dotsS[tid * 128 + col]);
    float s = 0.f;
    for (int col = 0; col < 128; ++col) s += expf(dotsS[tid * 128 + col] - m);
    float lse = m + logf(s);
    if (phase == 0) {
      logits[(size_t)(bh * NHASH + h) * TLEN + kid[tid]] = lse;
    } else {
      float wq = wbuf[(size_t)(bh * NHASH + h) * TLEN + kid[tid]];
      for (int col = 0; col < 128; ++col)
        P[tid * 128 + col] = (__bf16)(expf(dotsS[tid * 128 + col] - lse) * wq);
    }
  }
  if (phase == 0) return;
  __syncthreads();

  for (int dt = 0; dt < 8; ++dt) {            // bo = P @ V  (64x128), weighted
    v8f acc = zero8();
#pragma unroll
    for (int kk = 0; kk < 4; ++kk) {
      v16bf af = load_frag(P, qs, 128, kk * 32, lane);
      v16bf bf = load_frag(Vt, dt * 16, 128, kk * 32, lane);
      acc = wmma_bf16(af, bf, acc);
    }
#pragma unroll
    for (int r = 0; r < 8; ++r) {
      int rowl = qs + hi * 8 + r;
      int tok = kid[rowl];
      int d = dt * 16 + col0;
      atomicAdd(&attn[((size_t)bh * TLEN + tok) * DH + d], acc[r]);
    }
  }
}

// --------------------------------- cross-round logsumexp combination weights
__global__ void weights_kernel(const float* __restrict__ logits,
                               float* __restrict__ wbuf) {
  int i = blockIdx.x * blockDim.x + threadIdx.x;     // bh*TLEN + tok
  if (i >= BH * TLEN) return;
  int bh = i / TLEN, tok = i % TLEN;
  float l[NHASH];
  float m = -1e30f;
#pragma unroll
  for (int h = 0; h < NHASH; ++h) {
    l[h] = logits[(size_t)(bh * NHASH + h) * TLEN + tok];
    m = fmaxf(m, l[h]);
  }
  float s = 0.f;
#pragma unroll
  for (int h = 0; h < NHASH; ++h) s += expf(l[h] - m);
#pragma unroll
  for (int h = 0; h < NHASH; ++h)
    wbuf[(size_t)(bh * NHASH + h) * TLEN + tok] = expf(l[h] - m) / s;
}

// ---------------------------- (b,h,t,dh) f32 -> (b,t, h*dh) bf16 (merge heads)
__global__ void reorder_kernel(const float* __restrict__ attn,
                               __bf16* __restrict__ out) {
  size_t j = (size_t)blockIdx.x * blockDim.x + threadIdx.x;
  if (j >= (size_t)BATCH * TLEN * EMB) return;
  int b = (int)(j >> 23);
  int tok = (int)((j >> 10) & (TLEN - 1));
  int hh = (int)((j >> 7) & 7);
  int d = (int)(j & 127);
  out[j] = (__bf16)(attn[(((size_t)(b * HEADS + hh)) * TLEN + tok) * DH + d]);
}

// ============================================================================
extern "C" void kernel_launch(void* const* d_in, const int* in_sizes, int n_in,
                              void* d_out, int out_size, void* d_ws,
                              size_t ws_size, hipStream_t stream) {
  const float* x            = (const float*)d_in[0];
  const unsigned char* pmsk = (const unsigned char*)d_in[1];  // bool mask
  const float* rot          = (const float*)d_in[2];
  const float* Wqk          = (const float*)d_in[3];
  const float* Wv           = (const float*)d_in[4];
  const float* Wout         = (const float*)d_in[5];
  const float* bout         = (const float*)d_in[6];
  float* out                = (float*)d_out;

  char* ws = (char*)d_ws;
  size_t off = 0;
  auto alloc = [&](size_t bytes) -> void* {
    void* p = ws + off;
    off = (off + bytes + 255) & ~(size_t)255;
    return p;
  };

  const size_t MT = (size_t)BATCH * TLEN;  // 16384 rows
  __bf16* xb      = (__bf16*)alloc(MT * EMB * 2);
  __bf16* Wqkt    = (__bf16*)alloc((size_t)EMB * EMB * 2);   // [n][k] transposed
  __bf16* Wvt     = (__bf16*)alloc((size_t)EMB * EMB * 2);
  __bf16* Woutt   = (__bf16*)alloc((size_t)EMB * EMB * 2);
  float*  qkf     = (float*)alloc(MT * EMB * 4);
  float*  vf      = (float*)alloc(MT * EMB * 4);
  float*  normf   = (float*)alloc((size_t)BH * TLEN * 4);
  unsigned char* buckets = (unsigned char*)alloc((size_t)BH * NHASH * TLEN);
  int*    st      = (int*)alloc((size_t)BH * NHASH * TLEN * 4);
  float*  logits  = (float*)alloc((size_t)BH * NHASH * TLEN * 4);
  float*  wbuf    = (float*)alloc((size_t)BH * NHASH * TLEN * 4);
  float*  attn    = (float*)alloc((size_t)BH * TLEN * DH * 4);
  __bf16* attnb   = (__bf16*)alloc(MT * EMB * 2);

  // 1) bf16 conversions (weights transposed to [n][k] for async-friendly GEMM)
  {
    int n = (int)(MT * EMB);
    cvt_bf16_kernel<<<(n + 255) / 256, 256, 0, stream>>>(x, xb, n);
    int nw = EMB * EMB;
    cvt_bf16_t_kernel<<<(nw + 255) / 256, 256, 0, stream>>>(Wqk, Wqkt, EMB, EMB);
    cvt_bf16_t_kernel<<<(nw + 255) / 256, 256, 0, stream>>>(Wv, Wvt, EMB, EMB);
    cvt_bf16_t_kernel<<<(nw + 255) / 256, 256, 0, stream>>>(Wout, Woutt, EMB, EMB);
  }

  // 2) qk = x @ Wqk ; v = x @ Wv   (f32 accum)
  dim3 gg(EMB / 128, (unsigned)(MT / 128));
  gemm_bf16_kernel<<<gg, 256, 0, stream>>>(xb, Wqkt, qkf, nullptr,
                                           (int)MT, EMB, EMB);
  gemm_bf16_kernel<<<gg, 256, 0, stream>>>(xb, Wvt, vf, nullptr,
                                           (int)MT, EMB, EMB);

  // 3) LSH hashing + key-norm factors
  hash_kernel<<<BH * (TLEN / 64), 64, 64 * 1024, stream>>>(qkf, rot, buckets,
                                                           normf);

  // 4) stable counting sort per (bh, hash)
  sort_kernel<<<BH * NHASH, 128, 128 * 128 * 4 + 2 * 128 * 4, stream>>>(buckets,
                                                                        st);

  // 5) attention pass 0: per-round logsumexps
  size_t attn_smem = 64 * 128 * 2 + 2 * (128 * 128 * 2) + 64 * 128 * 4 +
                     64 * 128 * 2 + 128 * 4;
  lsh_attn_kernel<<<BH * NCHUNK, 128, attn_smem, stream>>>(
      qkf, vf, normf, st, pmsk, logits, wbuf, attn, 0);

  // 6) combination weights across hash rounds
  weights_kernel<<<(BH * TLEN + 255) / 256, 256, 0, stream>>>(logits, wbuf);

  // 7) attention pass 1: weighted PV accumulated into combined output
  hipMemsetAsync(attn, 0, (size_t)BH * TLEN * DH * 4, stream);
  lsh_attn_kernel<<<BH * NCHUNK, 128, attn_smem, stream>>>(
      qkf, vf, normf, st, pmsk, logits, wbuf, attn, 1);

  // 8) merge heads -> bf16, final projection with bias
  {
    size_t n = MT * EMB;
    reorder_kernel<<<(unsigned)((n + 255) / 256), 256, 0, stream>>>(attn, attnb);
  }
  gemm_bf16_kernel<<<gg, 256, 0, stream>>>(attnb, Woutt, out, bout,
                                           (int)MT, EMB, EMB);
}
